// TransformerBlock_81183471829224
// MI455X (gfx1250) — compile-verified, matrix operand order checked by reading, not run on hardware
//
#include <hip/hip_runtime.h>
#include <hip/hip_bf16.h>
#include <cstdint>

typedef __attribute__((ext_vector_type(16))) _Float16 v16h;
typedef __attribute__((ext_vector_type(8)))  _Float16 v8h;
typedef __attribute__((ext_vector_type(8)))  float    v8f;

#define DMODEL 1024
#define NHEADS 16
#define HDIM   64
#define SEQ    2048
#define BATCH  2
#define FDIM   4096
#define BSTOK  (BATCH*SEQ)   // 4096 rows
#define QKVLD  (3*DMODEL)    // packed q|k|v row stride

// ---------------------------------------------------------------------------
// 16-bit WMMA fragment load. CDNA5 16-bit A-fragment layout (ISA 7.12.2):
// lane l holds row M=l%16; element pair p holds K = (p<4?0:16)+(l<16?0:8)+(p&3)*2.
// => per lane, vector elements [0..8) are K in [c0, c0+8) and [8..16) are
// K in [c0+16, c0+24), c0 = (l<16 ? 0 : 8): two contiguous 16-byte loads.
// B-fragments mirror this with N=l%16, so B tiles are kept K-contiguous in LDS.
// ---------------------------------------------------------------------------
__device__ __forceinline__ v16h load_frag(const _Float16* base, int stride,
                                          int row, int k0, int lane) {
  const int c0 = (lane < 16) ? 0 : 8;
  const _Float16* p = base + row * stride + k0 + c0;
  v8h lo = *(const v8h*)(p);
  v8h hi = *(const v8h*)(p + 16);
  v16h r;
#pragma unroll
  for (int i = 0; i < 8; ++i) { r[i] = lo[i]; r[i + 8] = hi[i]; }
  return r;
}

__device__ __forceinline__ v8f wmma16(v16h a, v16h b, v8f c) {
  return __builtin_amdgcn_wmma_f32_16x16x32_f16(false, a, false, b,
                                                (short)0, c, false, false);
}

// Async global->LDS copy of 16 bytes (VFLAT GLOBAL_LOAD_ASYNC_TO_LDS_B128).
// Tracked on ASYNCcnt; fenced with s_wait_asynccnt + workgroup barrier.
__device__ __forceinline__ void async_copy_b128(const _Float16* gsrc, _Float16* ldst) {
  unsigned lds = (unsigned)reinterpret_cast<uintptr_t>((void*)ldst);
  unsigned long long ga = (unsigned long long)reinterpret_cast<uintptr_t>((const void*)gsrc);
  asm volatile("global_load_async_to_lds_b128 %0, %1, off"
               :: "v"(lds), "v"(ga) : "memory");
}
__device__ __forceinline__ void async_wait0() {
  asm volatile("s_wait_asynccnt 0" ::: "memory");
}

__device__ __forceinline__ float gelu_tanh(float x) {
  const float k0 = 0.7978845608028654f;  // sqrt(2/pi)
  float t = tanhf(k0 * (x + 0.044715f * x * x * x));
  return 0.5f * x * (1.0f + t);
}

// ---------------------------------------------------------------------------
// fp32 -> fp16 conversion with output leading-dimension (for q|k|v packing)
// ---------------------------------------------------------------------------
__global__ __launch_bounds__(256) void cvt_f16_kernel(const float* __restrict__ src,
                                                      _Float16* __restrict__ dst,
                                                      int n, int cols, int ldd) {
  int i = (blockIdx.x * 256 + threadIdx.x) * 4;
  if (i < n) {
    float4 v = *(const float4*)(src + i);
    int r = i / cols, c = i - r * cols;
    _Float16* d = dst + (size_t)r * ldd + c;
    d[0] = (_Float16)v.x;
    d[1] = (_Float16)v.y;
    d[2] = (_Float16)v.z;
    d[3] = (_Float16)v.w;
  }
}

// ---------------------------------------------------------------------------
// srmsnorm: y = x * sqrt(D) / max(||x||, eps), emitted as f16. One block/row.
// ---------------------------------------------------------------------------
__global__ __launch_bounds__(256) void rmsnorm_kernel(const float* __restrict__ x,
                                                      _Float16* __restrict__ out) {
  __shared__ float red[8];
  const int row = blockIdx.x;
  const int tid = threadIdx.x, lane = tid & 31, w = tid >> 5;
  const float* xr = x + (size_t)row * DMODEL;
  float4 xv = ((const float4*)xr)[tid];
  float s = xv.x * xv.x + xv.y * xv.y + xv.z * xv.z + xv.w * xv.w;
#pragma unroll
  for (int off = 16; off > 0; off >>= 1) s += __shfl_xor(s, off, 32);
  if (lane == 0) red[w] = s;
  __syncthreads();
  float tot = 0.f;
#pragma unroll
  for (int i = 0; i < 8; ++i) tot += red[i];
  float norm = sqrtf(tot);
  float sc = 32.0f / fmaxf(norm, 1e-12f);  // sqrt(1024) = 32
  _Float16* o = out + (size_t)row * DMODEL + tid * 4;
  o[0] = (_Float16)(xv.x * sc);
  o[1] = (_Float16)(xv.y * sc);
  o[2] = (_Float16)(xv.z * sc);
  o[3] = (_Float16)(xv.w * sc);
}

// ---------------------------------------------------------------------------
// WMMA GEMM: C[M,N] = op(A[M,K] @ B[K,N]) (+ residual), A/B f16 row-major.
// Block tile 256x128x32; 8 waves in 4x2; wave tile 64x64 (16 WMMA / k-step).
// Double-buffered LDS; A tile staged with async global->LDS (ASYNCcnt),
// B tile staged through registers (needs the K-contiguous transpose).
// ---------------------------------------------------------------------------
template <bool GELU, bool RESID, bool OUT_F32, bool OUT_F16>
__global__ __launch_bounds__(256)
void gemm_wmma_kernel(const _Float16* __restrict__ A, const _Float16* __restrict__ Bm,
                      const float* __restrict__ Res, float* __restrict__ Cf,
                      _Float16* __restrict__ Ch, int M, int N, int K) {
  __shared__ __align__(16) _Float16 sA[2][256 * 40];  // [m][k], stride 40
  __shared__ __align__(16) _Float16 sB[2][128 * 40];  // [n][k], stride 40
  const int tid = threadIdx.x, lane = tid & 31, w = tid >> 5;
  const int wm = w >> 1, wn = w & 1;                  // 4x2 wave grid
  const int row0 = blockIdx.x * 256, col0 = blockIdx.y * 128;

  v8f acc[4][4];
#pragma unroll
  for (int a = 0; a < 4; ++a)
#pragma unroll
    for (int b = 0; b < 4; ++b) acc[a][b] = (v8f){};

  v8h breg[2];
  auto issueA = [&](int kk, int bb) {
#pragma unroll
    for (int i = 0; i < 4; ++i) {
      int f = (i * 256 + tid) * 8;
      int r = f >> 5, kq = f & 31;
      async_copy_b128(A + (size_t)(row0 + r) * K + kk + kq, &sA[bb][r * 40 + kq]);
    }
  };
  auto loadB = [&](int kk) {
#pragma unroll
    for (int i = 0; i < 2; ++i) {
      int f = (i * 256 + tid) * 8;
      int kx = f >> 7, n = f & 127;
      breg[i] = *(const v8h*)(Bm + (size_t)(kk + kx) * N + col0 + n);
    }
  };
  auto storeB = [&](int bb) {
#pragma unroll
    for (int i = 0; i < 2; ++i) {
      int f = (i * 256 + tid) * 8;
      int kx = f >> 7, n = f & 127;
#pragma unroll
      for (int j = 0; j < 8; ++j) sB[bb][(n + j) * 40 + kx] = breg[i][j];
    }
  };

  issueA(0, 0);
  loadB(0);
  const int T = K / 32;
  for (int t = 0; t < T; ++t) {
    const int bb = t & 1;
    storeB(bb);
    async_wait0();        // tile t's A data resident in sA[bb]
    __syncthreads();
    if (t + 1 < T) {      // overlap tile t+1 staging with tile t compute
      issueA((t + 1) * 32, bb ^ 1);
      loadB((t + 1) * 32);
    }
    v16h af[4], bf[4];
#pragma unroll
    for (int tm = 0; tm < 4; ++tm)
      af[tm] = load_frag(&sA[bb][0], 40, wm * 64 + tm * 16 + (lane & 15), 0, lane);
#pragma unroll
    for (int tn = 0; tn < 4; ++tn)
      bf[tn] = load_frag(&sB[bb][0], 40, wn * 64 + tn * 16 + (lane & 15), 0, lane);
#pragma unroll
    for (int tm = 0; tm < 4; ++tm)
#pragma unroll
      for (int tn = 0; tn < 4; ++tn)
        acc[tm][tn] = wmma16(af[tm], bf[tn], acc[tm][tn]);
    __syncthreads();
  }

  // epilogue: C-fragment layout (lanes 0-15 -> M=v, lanes 16-31 -> M=v+8)
  const int mh = (lane >= 16) ? 8 : 0, nc = lane & 15;
#pragma unroll
  for (int tm = 0; tm < 4; ++tm)
#pragma unroll
    for (int tn = 0; tn < 4; ++tn)
#pragma unroll
      for (int v = 0; v < 8; ++v) {
        int r = row0 + wm * 64 + tm * 16 + v + mh;
        int c = col0 + wn * 64 + tn * 16 + nc;
        float x = acc[tm][tn][v];
        if (GELU) x = gelu_tanh(x);
        if (RESID) x += Res[(size_t)r * N + c];
        if (OUT_F32) Cf[(size_t)r * N + c] = x;
        if (OUT_F16) Ch[(size_t)r * N + c] = (_Float16)x;
      }
}

// ---------------------------------------------------------------------------
// Causal flash attention over packed qkv [BS, 3D]. Grid (SEQ/64, B*H),
// 128 threads = 4 waves; each wave owns 16 queries; 64-key K/V tiles shared
// in LDS (K tile via async copy, V tile transposed through registers).
// ---------------------------------------------------------------------------
__global__ __launch_bounds__(128)
void attention_kernel(const _Float16* __restrict__ qkv, _Float16* __restrict__ o) {
  __shared__ __align__(16) _Float16 sK[64 * 72];       // [key][hd], stride 72
  __shared__ __align__(16) _Float16 sV[64 * 72];       // [hd][key] (transposed)
  __shared__ __align__(16) _Float16 sP[4 * 16 * 72];   // per-wave P staging
  const int tid = threadIdx.x, lane = tid & 31, w = tid >> 5;
  const int qblk = blockIdx.x, bh = blockIdx.y;
  const int b = bh >> 4, h = bh & (NHEADS - 1);
  const int q0 = qblk * 64 + w * 16;
  const size_t rowB = (size_t)b * SEQ;
  const size_t colQ = (size_t)h * HDIM;

  v16h aq[2];
  {
    const _Float16* qp = qkv + (rowB + q0) * QKVLD + colQ;
    aq[0] = load_frag(qp, QKVLD, lane & 15, 0, lane);
    aq[1] = load_frag(qp, QKVLD, lane & 15, 32, lane);
  }

  v8f acc[4];
  float Mrow[8], Lrow[8];
#pragma unroll
  for (int i = 0; i < 4; ++i) acc[i] = (v8f){};
#pragma unroll
  for (int i = 0; i < 8; ++i) { Mrow[i] = -1e30f; Lrow[i] = 0.f; }

  const int kend = qblk * 64 + 64;  // causal: keys < kend needed by this block
  for (int j = 0; j < kend; j += 64) {
    // cooperative K/V tile load (K: async straight copy, V: transposed)
#pragma unroll
    for (int i = 0; i < 4; ++i) {
      int f = (i * 128 + tid) * 8;
      int kk = f >> 6, d = f & 63;
      const _Float16* krow = qkv + (rowB + j + kk) * QKVLD + DMODEL + colQ;
      async_copy_b128(krow + d, sK + kk * 72 + d);
      v8h vv = *(const v8h*)(qkv + (rowB + j + kk) * QKVLD + 2 * DMODEL + colQ + d);
#pragma unroll
      for (int t = 0; t < 8; ++t) sV[(d + t) * 72 + kk] = vv[t];
    }
    async_wait0();
    __syncthreads();

    if (j <= q0 + 15) {  // wave-uniform skip of fully-masked tiles
      // ---- scores: 4 key sub-tiles of 16, K-dim = hd = 64 (2 WMMA chunks)
      v8f s[4];
#pragma unroll
      for (int t = 0; t < 4; ++t) {
        v16h bk0 = load_frag(sK, 72, t * 16 + (lane & 15), 0, lane);
        v16h bk1 = load_frag(sK, 72, t * 16 + (lane & 15), 32, lane);
        v8f z = (v8f){};
        z = wmma16(aq[0], bk0, z);
        z = wmma16(aq[1], bk1, z);
        s[t] = z;
      }
      // ---- scale + causal mask + row max
      const int mh = (lane >= 16) ? 8 : 0, nc = lane & 15;
      float mloc[8];
#pragma unroll
      for (int i = 0; i < 8; ++i) mloc[i] = -1e30f;
#pragma unroll
      for (int t = 0; t < 4; ++t)
#pragma unroll
        for (int vv = 0; vv < 8; ++vv) {
          float val = s[t][vv] * 0.125f;  // 1/sqrt(64)
          bool ok = (j + t * 16 + nc) <= (q0 + mh + vv);
          val = ok ? val : -1e30f;
          s[t][vv] = val;
          mloc[vv] = fmaxf(mloc[vv], val);
        }
#pragma unroll
      for (int vv = 0; vv < 8; ++vv)
#pragma unroll
        for (int off = 8; off > 0; off >>= 1)
          mloc[vv] = fmaxf(mloc[vv], __shfl_xor(mloc[vv], off, 16));
      // ---- online softmax update
      float alpha[8], rs[8];
#pragma unroll
      for (int vv = 0; vv < 8; ++vv) {
        float mn = fmaxf(Mrow[vv], mloc[vv]);
        alpha[vv] = __expf(Mrow[vv] - mn);
        Mrow[vv] = mn;
        rs[vv] = 0.f;
      }
#pragma unroll
      for (int t = 0; t < 4; ++t)
#pragma unroll
        for (int vv = 0; vv < 8; ++vv) {
          float p = __expf(s[t][vv] - Mrow[vv]);
          s[t][vv] = p;
          rs[vv] += p;
        }
#pragma unroll
      for (int vv = 0; vv < 8; ++vv) {
#pragma unroll
        for (int off = 8; off > 0; off >>= 1) rs[vv] += __shfl_xor(rs[vv], off, 16);
        Lrow[vv] = Lrow[vv] * alpha[vv] + rs[vv];
      }
#pragma unroll
      for (int ct = 0; ct < 4; ++ct)
#pragma unroll
        for (int vv = 0; vv < 8; ++vv) acc[ct][vv] *= alpha[vv];
      // ---- transpose P (C-layout -> A-layout) through per-wave LDS scratch
      _Float16* pw = sP + w * 16 * 72;
#pragma unroll
      for (int t = 0; t < 4; ++t)
#pragma unroll
        for (int vv = 0; vv < 8; ++vv)
          pw[(vv + mh) * 72 + t * 16 + nc] = (_Float16)s[t][vv];
      // same-wave DS ops are in-order; backend inserts the dscnt wait
      v16h ap0 = load_frag(pw, 72, lane & 15, 0, lane);
      v16h ap1 = load_frag(pw, 72, lane & 15, 32, lane);
      // ---- PV: out[16 x 64] over 4 hd column tiles, K-dim = 64 keys
#pragma unroll
      for (int ct = 0; ct < 4; ++ct) {
        v16h bv0 = load_frag(sV, 72, ct * 16 + (lane & 15), 0, lane);
        v16h bv1 = load_frag(sV, 72, ct * 16 + (lane & 15), 32, lane);
        acc[ct] = wmma16(ap0, bv0, acc[ct]);
        acc[ct] = wmma16(ap1, bv1, acc[ct]);
      }
    }
    __syncthreads();
  }

  // ---- normalize and store f16 at [b, q, h*64 + d]
  _Float16* op = o + (rowB + q0) * DMODEL + colQ;
  const int mh = (lane >= 16) ? 8 : 0, nc = lane & 15;
#pragma unroll
  for (int ct = 0; ct < 4; ++ct)
#pragma unroll
    for (int vv = 0; vv < 8; ++vv)
      op[(size_t)(vv + mh) * DMODEL + ct * 16 + nc] =
          (_Float16)(acc[ct][vv] / Lrow[vv]);
}

// ---------------------------------------------------------------------------
// Host launcher
// ---------------------------------------------------------------------------
extern "C" void kernel_launch(void* const* d_in, const int* in_sizes, int n_in,
                              void* d_out, int out_size, void* d_ws, size_t ws_size,
                              hipStream_t stream) {
  (void)in_sizes; (void)n_in; (void)out_size; (void)ws_size;
  const float* x     = (const float*)d_in[0];
  const float* wq    = (const float*)d_in[1];
  const float* wk    = (const float*)d_in[2];
  const float* wv    = (const float*)d_in[3];
  const float* wo    = (const float*)d_in[4];
  const float* w_in  = (const float*)d_in[5];
  const float* w_out = (const float*)d_in[6];
  float* out = (float*)d_out;

  size_t off = 0;
  auto alloc = [&](size_t bytes) -> void* {
    void* p = (char*)d_ws + off;
    off += (bytes + 255) & ~(size_t)255;
    return p;
  };
  _Float16* wqkv = (_Float16*)alloc((size_t)DMODEL * QKVLD * 2);  // packed [D, 3D]
  _Float16* woh  = (_Float16*)alloc((size_t)DMODEL * DMODEL * 2);
  _Float16* wih  = (_Float16*)alloc((size_t)DMODEL * FDIM * 2);
  _Float16* wouh = (_Float16*)alloc((size_t)FDIM * DMODEL * 2);
  _Float16* xn   = (_Float16*)alloc((size_t)BSTOK * DMODEL * 2);
  _Float16* qkv  = (_Float16*)alloc((size_t)BSTOK * QKVLD * 2);  // packed q|k|v
  _Float16* attn = (_Float16*)alloc((size_t)BSTOK * DMODEL * 2);
  _Float16* hn   = (_Float16*)alloc((size_t)BSTOK * DMODEL * 2);
  _Float16* gh   = (_Float16*)alloc((size_t)BSTOK * FDIM * 2);
  float*    hf   = (float*)   alloc((size_t)BSTOK * DMODEL * 4);

  auto cvt = [&](const float* s, _Float16* d, int rows, int cols, int ldd) {
    int n = rows * cols;
    cvt_f16_kernel<<<(n / 4 + 255) / 256, 256, 0, stream>>>(s, d, n, cols, ldd);
  };
  cvt(wq, wqkv + 0 * DMODEL, DMODEL, DMODEL, QKVLD);
  cvt(wk, wqkv + 1 * DMODEL, DMODEL, DMODEL, QKVLD);
  cvt(wv, wqkv + 2 * DMODEL, DMODEL, DMODEL, QKVLD);
  cvt(wo, woh, DMODEL, DMODEL, DMODEL);
  cvt(w_in, wih, DMODEL, FDIM, FDIM);
  cvt(w_out, wouh, FDIM, DMODEL, DMODEL);

  // n1 = srmsnorm(x)
  rmsnorm_kernel<<<BSTOK, 256, 0, stream>>>(x, xn);

  // fused q|k|v projection: [BS,D] @ [D,3D] -> [BS,3D] (f16)
  gemm_wmma_kernel<false, false, false, true>
      <<<dim3(BSTOK / 256, QKVLD / 128), 256, 0, stream>>>(
          xn, wqkv, nullptr, nullptr, qkv, BSTOK, QKVLD, DMODEL);

  // causal attention
  attention_kernel<<<dim3(SEQ / 64, BATCH * NHEADS), 128, 0, stream>>>(qkv, attn);

  // h = attn @ wo + x  (f32)
  gemm_wmma_kernel<false, true, true, false>
      <<<dim3(BSTOK / 256, DMODEL / 128), 256, 0, stream>>>(
          attn, woh, x, hf, nullptr, BSTOK, DMODEL, DMODEL);

  // n2 = srmsnorm(h)
  rmsnorm_kernel<<<BSTOK, 256, 0, stream>>>(hf, hn);

  // g = gelu(n2 @ w_in)  (f16)
  gemm_wmma_kernel<true, false, false, true>
      <<<dim3(BSTOK / 256, FDIM / 128), 256, 0, stream>>>(
          hn, wih, nullptr, nullptr, gh, BSTOK, FDIM, DMODEL);

  // out = g @ w_out + h  (f32)
  gemm_wmma_kernel<false, true, true, false>
      <<<dim3(BSTOK / 256, DMODEL / 128), 256, 0, stream>>>(
          gh, wouh, hf, out, nullptr, BSTOK, DMODEL, FDIM);
}
